// gatv2_block_39393440039109
// MI455X (gfx1250) — matrix operand order repaired and checked by reading, not compiled, not run on hardware
//
#include <hip/hip_runtime.h>

#define IN_CH      128
#define HEADS      8
#define OUT_CH     32
#define CH         256           // HEADS*OUT_CH
#define XLR_STRIDE 512           // [xl | xr] per node
#define NEG_SLOPE  0.2f
#define GN_EPS     1e-5f

typedef __attribute__((ext_vector_type(16))) __bf16 v16bf;
typedef __attribute__((ext_vector_type(8)))  __bf16 v8bf;
typedef __attribute__((ext_vector_type(8)))  float  v8f;

// ---------- helpers ----------

static __device__ __forceinline__ __bf16 bf16_from_bits(unsigned short u) {
    return __builtin_bit_cast(__bf16, u);
}

// split fp32 into bf16 hi + bf16 residual (truncating split; residual captures low bits)
static __device__ __forceinline__ void split_f32(float f, __bf16& hi, __bf16& lo) {
    unsigned u = __float_as_uint(f);
    unsigned short h = (unsigned short)(u >> 16);
    float hif = __uint_as_float((unsigned)h << 16);
    float r = f - hif;
    unsigned short l = (unsigned short)(__float_as_uint(r) >> 16);
    hi = bf16_from_bits(h);
    lo = bf16_from_bits(l);
}

// monotone float -> int key so signed-int atomicMax implements float max
static __device__ __forceinline__ int key_encode(float f) {
    int i = __float_as_int(f);
    return (i < 0) ? (i ^ 0x7fffffff) : i;
}
static __device__ __forceinline__ float key_decode(int k) {
    int i = (k < 0) ? (k ^ 0x7fffffff) : k;
    return __int_as_float(i);
}

static __device__ __forceinline__ float lrelu_dot(float4 a, float4 b, float4 w) {
    float acc = 0.f, v;
    v = a.x + b.x; acc += w.x * (v > 0.f ? v : NEG_SLOPE * v);
    v = a.y + b.y; acc += w.y * (v > 0.f ? v : NEG_SLOPE * v);
    v = a.z + b.z; acc += w.z * (v > 0.f ? v : NEG_SLOPE * v);
    v = a.w + b.w; acc += w.w * (v > 0.f ? v : NEG_SLOPE * v);
    return acc;
}

// ---------- init fills ----------

__global__ void k_fill_f32(float* __restrict__ p, float v, int n) {
    int i = blockIdx.x * blockDim.x + threadIdx.x;
    if (i < n) p[i] = v;
}
__global__ void k_fill_key(int* __restrict__ p, float f, int n) {
    int i = blockIdx.x * blockDim.x + threadIdx.x;
    if (i < n) p[i] = key_encode(f);
}

// ---------- precision-split conversion (done ONCE, removes VALU from GEMM loop) ----

// x [N,128] fp32 -> xhi/xlo [N,128] bf16 planes (row-major)
__global__ __launch_bounds__(256) void k_conv_x(
    const float* __restrict__ x, __bf16* __restrict__ xhi,
    __bf16* __restrict__ xlo, int n)
{
    int i = blockIdx.x * blockDim.x + threadIdx.x;
    if (i >= n) return;
    __bf16 h, l; split_f32(x[i], h, l);
    xhi[i] = h; xlo[i] = l;
}

// W combined [128, 512] fp32 (Wl | Wr) -> TRANSPOSED bf16 planes wt[col][k], col in [0,512)
__global__ __launch_bounds__(128) void k_conv_w(
    const float* __restrict__ Wl, const float* __restrict__ Wr,
    __bf16* __restrict__ wthi, __bf16* __restrict__ wtlo)
{
    const int c = blockIdx.x;        // 512 output columns
    const int k = threadIdx.x;       // 128 k values
    const float v = (c < CH) ? Wl[(size_t)k * CH + c] : Wr[(size_t)k * CH + (c - CH)];
    __bf16 h, l; split_f32(v, h, l);
    wthi[(size_t)c * IN_CH + k] = h;
    wtlo[(size_t)c * IN_CH + k] = l;
}

// ---------- fused linear: xlr[n, 0:256]=x@Wl+bl, xlr[n,256:512]=x@Wr+br ----------
// one wave -> 16 rows x 64 cols (4 WMMA tiles); bf16x3 emulation of fp32 GEMM.
// Operands are pre-split bf16 planes: the K-loop is pure loads + WMMA.
__global__ __launch_bounds__(256) void k_gemm(
    const __bf16* __restrict__ xhi, const __bf16* __restrict__ xlo,
    const __bf16* __restrict__ wthi, const __bf16* __restrict__ wtlo,
    const float* __restrict__ bl, const float* __restrict__ br,
    float* __restrict__ xlr, int nNodes)
{
    const int lane = threadIdx.x & 31;
    const int wave = threadIdx.x >> 5;          // 0..7 -> 64-col group
    const int rowBase = blockIdx.x * 16;
    if (rowBase >= nNodes) return;
    const int colGroup = wave * 64;             // 0..448 (global col in [0,512))
    const int l15  = lane & 15;
    const int hi16 = lane >> 4;                 // lane half
    const int row  = rowBase + l15;

    v8f acc[4] = {};

#pragma unroll
    for (int k0 = 0; k0 < IN_CH; k0 += 32) {
        // A tile 16x32 bf16: lanes 0-15 K=k0..k0+15, lanes 16-31 K=k0+16..k0+31 (matches B)
        const __bf16* xph = xhi + (size_t)row * IN_CH + k0 + hi16 * 16;
        const __bf16* xpl = xlo + (size_t)row * IN_CH + k0 + hi16 * 16;
        const v16bf ahi = __builtin_shufflevector(
            *(const v8bf*)xph, *(const v8bf*)(xph + 8),
            0,1,2,3,4,5,6,7,8,9,10,11,12,13,14,15);
        const v16bf alo = __builtin_shufflevector(
            *(const v8bf*)xpl, *(const v8bf*)(xpl + 8),
            0,1,2,3,4,5,6,7,8,9,10,11,12,13,14,15);
#pragma unroll
        for (int t = 0; t < 4; ++t) {
            // B tile 32x16 bf16: lanes 0-15 K=k0..k0+15, lanes 16-31 K=k0+16..k0+31
            const int col = colGroup + t * 16 + l15;
            const size_t wo = (size_t)col * IN_CH + k0 + hi16 * 16;
            const v16bf bhi = *(const v16bf*)(wthi + wo);   // 32 contiguous bytes
            const v16bf blo = *(const v16bf*)(wtlo + wo);
            acc[t] = __builtin_amdgcn_wmma_f32_16x16x32_bf16(
                false, ahi, false, bhi, (short)0, acc[t], false, false);
            acc[t] = __builtin_amdgcn_wmma_f32_16x16x32_bf16(
                false, ahi, false, blo, (short)0, acc[t], false, false);
            acc[t] = __builtin_amdgcn_wmma_f32_16x16x32_bf16(
                false, alo, false, bhi, (short)0, acc[t], false, false);
        }
    }

    // D layout: VGPR r -> M = r + (lane>=16 ? 8 : 0), N = lane&15
    const int mrow = rowBase + hi16 * 8;
#pragma unroll
    for (int t = 0; t < 4; ++t) {
        const int col = colGroup + t * 16 + l15;
        const float bias = (col < CH) ? bl[col] : br[col - CH];
#pragma unroll
        for (int r = 0; r < 8; ++r)
            xlr[(size_t)(mrow + r) * XLR_STRIDE + col] = acc[t][r] + bias;
    }
}

// ---------- edge pass 1: logits + segment max (one wave per edge) ----------
__global__ __launch_bounds__(256) void k_edge_logits(
    const int* __restrict__ ei, const float* __restrict__ xlr,
    const float* __restrict__ att, float* __restrict__ logits,
    int* __restrict__ mkey, int E, int N)
{
    const int lane = threadIdx.x & 31;
    const int e = blockIdx.x * 8 + (threadIdx.x >> 5);
    if (e >= E + N) return;
    int s, d;
    if (e < E) { s = ei[e]; d = ei[E + e]; if (s == d) return; }   // masked self-loop
    else       { s = e - E; d = s; }                               // appended self-loop

    const float4* xs = (const float4*)(xlr + (size_t)s * XLR_STRIDE + lane * 8);
    const float4* xd = (const float4*)(xlr + (size_t)d * XLR_STRIDE + CH + lane * 8);
    const float4* ap = (const float4*)(att + lane * 8);
    float partial = lrelu_dot(xs[0], xd[0], ap[0]) + lrelu_dot(xs[1], xd[1], ap[1]);
    // reduce the 4 lanes covering one head (8 ch each)
    partial += __shfl_xor(partial, 1, 32);
    partial += __shfl_xor(partial, 2, 32);
    if ((lane & 3) == 0) {
        const int h = lane >> 2;
        logits[(size_t)e * HEADS + h] = partial;
        atomicMax(mkey + (size_t)d * HEADS + h, key_encode(partial));
    }
}

// ---------- edge pass 2: alpha = exp(logit - m[d]); denom += alpha ----------
__global__ __launch_bounds__(256) void k_edge_alpha(
    const int* __restrict__ ei, float* __restrict__ logits,
    const int* __restrict__ mkey, float* __restrict__ denom, int E, int N)
{
    const int idx = blockIdx.x * blockDim.x + threadIdx.x;
    if (idx >= (E + N) * HEADS) return;
    const int e = idx >> 3, h = idx & 7;
    int d;
    if (e < E) { int s = ei[e]; d = ei[E + e]; if (s == d) return; }
    else       d = e - E;
    const float m = key_decode(mkey[(size_t)d * HEADS + h]);
    const float a = __expf(logits[idx] - m);
    logits[idx] = a;                                   // reuse buffer as alpha
    atomicAdd(denom + (size_t)d * HEADS + h, a);
}

// ---------- edge pass 3: out[d] += xl[s] * alpha/denom (one wave per edge) ----------
__global__ __launch_bounds__(256) void k_edge_scatter(
    const int* __restrict__ ei, const float* __restrict__ xlr,
    const float* __restrict__ alpha, const float* __restrict__ denom,
    float* __restrict__ out_acc, int E, int N)
{
    const int lane = threadIdx.x & 31;
    const int e = blockIdx.x * 8 + (threadIdx.x >> 5);
    if (e >= E + N) return;
    int s, d;
    if (e < E) { s = ei[e]; d = ei[E + e]; if (s == d) return; }
    else       { s = e - E; d = s; }
    const int h = lane >> 2;
    const float w = alpha[(size_t)e * HEADS + h] / denom[(size_t)d * HEADS + h];
    const float4* xs = (const float4*)(xlr + (size_t)s * XLR_STRIDE + lane * 8);
    const float4 v0 = xs[0], v1 = xs[1];
    float* op = out_acc + (size_t)d * CH + lane * 8;
    atomicAdd(op + 0, v0.x * w); atomicAdd(op + 1, v0.y * w);
    atomicAdd(op + 2, v0.z * w); atomicAdd(op + 3, v0.w * w);
    atomicAdd(op + 4, v1.x * w); atomicAdd(op + 5, v1.y * w);
    atomicAdd(op + 6, v1.z * w); atomicAdd(op + 7, v1.w * w);
}

// ---------- GraphNorm: column sums ----------
__global__ __launch_bounds__(256) void k_col_stats(
    const float* __restrict__ acc, const float* __restrict__ bias_out,
    float* __restrict__ colsum, float* __restrict__ colsq, int N)
{
    const int c = threadIdx.x;                  // 256 columns
    const float b = bias_out[c];
    float s = 0.f, s2 = 0.f;
    for (int r = blockIdx.x; r < N; r += gridDim.x) {
        const float v = acc[(size_t)r * CH + c] + b;
        s += v; s2 += v * v;
    }
    atomicAdd(&colsum[c], s);
    atomicAdd(&colsq[c], s2);
}

__global__ void k_col_params(
    const float* __restrict__ colsum, const float* __restrict__ colsq,
    const float* __restrict__ gnw, const float* __restrict__ gnb,
    const float* __restrict__ gms, float* __restrict__ scale,
    float* __restrict__ shift, int N)
{
    const int c = threadIdx.x;
    const float invN = 1.0f / (float)N;
    const float mean = colsum[c] * invN;
    const float m2   = colsq[c] * invN;
    const float sm   = gms[c] * mean;           // gn_mean_scale * mean
    // var = E[(v - sm)^2] = E[v^2] - 2*sm*E[v] + sm^2
    const float var  = m2 - 2.f * sm * mean + sm * sm;
    const float rs   = gnw[c] * rsqrtf(var + GN_EPS);
    scale[c] = rs;
    shift[c] = gnb[c] - rs * sm;
}

__global__ __launch_bounds__(256) void k_normalize(
    const float* __restrict__ acc, const float* __restrict__ bias_out,
    const float* __restrict__ scale, const float* __restrict__ shift,
    float* __restrict__ out, int total)
{
    const int i = blockIdx.x * blockDim.x + threadIdx.x;
    if (i >= total) return;
    const int c = i & (CH - 1);
    const float v = acc[i] + bias_out[c];
    const float r = scale[c] * v + shift[c];
    out[i] = r > 0.f ? r : 0.f;
}

// ---------- launcher ----------
extern "C" void kernel_launch(void* const* d_in, const int* in_sizes, int n_in,
                              void* d_out, int out_size, void* d_ws, size_t ws_size,
                              hipStream_t stream) {
    (void)n_in; (void)out_size; (void)ws_size;
    const float* x        = (const float*)d_in[0];
    const int*   ei       = (const int*)  d_in[1];
    const float* Wl       = (const float*)d_in[2];
    const float* bl       = (const float*)d_in[3];
    const float* Wr       = (const float*)d_in[4];
    const float* br       = (const float*)d_in[5];
    const float* att      = (const float*)d_in[6];
    const float* bias_out = (const float*)d_in[7];
    const float* gnw      = (const float*)d_in[8];
    const float* gnb      = (const float*)d_in[9];
    const float* gms      = (const float*)d_in[10];
    float* out = (float*)d_out;

    const int N  = in_sizes[0] / IN_CH;   // 20000
    const int E  = in_sizes[1] / 2;       // 320000
    const int ET = E + N;                 // edges + appended self-loops

    // workspace layout (in float units)
    float* ws = (float*)d_ws;
    size_t off = 0;
    float* xlr    = ws + off; off += (size_t)N * XLR_STRIDE;   // 10.24M
    float* logits = ws + off; off += (size_t)ET * HEADS;       // 2.72M (reused as alpha)
    int*   mkey   = (int*)(ws + off); off += (size_t)N * HEADS;
    float* denom  = ws + off; off += (size_t)N * HEADS;        // | contiguous zero-fill
    float* outacc = ws + off; off += (size_t)N * CH;           // |
    float* colsum = ws + off; off += CH;                       // |
    float* colsq  = ws + off; off += CH;                       // |
    float* scale  = ws + off; off += CH;
    float* shift  = ws + off; off += CH;
    __bf16* xhi   = (__bf16*)(ws + off); off += (size_t)N * IN_CH / 2;
    __bf16* xlo   = (__bf16*)(ws + off); off += (size_t)N * IN_CH / 2;
    __bf16* wthi  = (__bf16*)(ws + off); off += (size_t)XLR_STRIDE * IN_CH / 2;
    __bf16* wtlo  = (__bf16*)(ws + off); off += (size_t)XLR_STRIDE * IN_CH / 2;

    // init
    const int nkey = N * HEADS;
    k_fill_key<<<(nkey + 255) / 256, 256, 0, stream>>>(mkey, -3.0e38f, nkey);
    const int nz = N * HEADS + N * CH + 2 * CH;   // denom + outacc + colsum + colsq
    k_fill_f32<<<(nz + 255) / 256, 256, 0, stream>>>(denom, 0.f, nz);

    // precision split (once)
    const int nx = N * IN_CH;
    k_conv_x<<<(nx + 255) / 256, 256, 0, stream>>>(x, xhi, xlo, nx);
    k_conv_w<<<XLR_STRIDE, IN_CH, 0, stream>>>(Wl, Wr, wthi, wtlo);

    // pipeline
    k_gemm<<<(N + 15) / 16, 256, 0, stream>>>(xhi, xlo, wthi, wtlo, bl, br, xlr, N);
    k_edge_logits<<<(ET + 7) / 8, 256, 0, stream>>>(ei, xlr, att, logits, mkey, E, N);
    const int na = ET * HEADS;
    k_edge_alpha<<<(na + 255) / 256, 256, 0, stream>>>(ei, logits, mkey, denom, E, N);
    k_edge_scatter<<<(ET + 7) / 8, 256, 0, stream>>>(ei, xlr, logits, denom, outacc, E, N);
    k_col_stats<<<64, CH, 0, stream>>>(outacc, bias_out, colsum, colsq, N);
    k_col_params<<<1, CH, 0, stream>>>(colsum, colsq, gnw, gnb, gms, scale, shift, N);
    const int nt = N * CH;
    k_normalize<<<(nt + 255) / 256, 256, 0, stream>>>(outacc, bias_out, scale, shift, out, nt);
}
